// GatedDeltaNetBackbone_85246510891202
// MI455X (gfx1250) — compile-verified, matrix operand order checked by reading, not run on hardware
//
#include <hip/hip_runtime.h>
#include <hip/hip_bf16.h>
#include <math.h>

// ---------------------------------------------------------------------------
// GatedDeltaNet backbone for MI455X (gfx1250, wave32, WMMA).
// Heavy linears -> v_wmma_f32_16x16x32_f16 (f16 operands, fp32 accumulate),
// with double-buffered async global->LDS staging (ASYNCcnt) for the A tiles.
// Residual stream / scan state / norms stay fp32.
// ---------------------------------------------------------------------------

typedef _Float16 v8h  __attribute__((ext_vector_type(8)));
typedef _Float16 v16h __attribute__((ext_vector_type(16)));
typedef float    v8f  __attribute__((ext_vector_type(8)));

#define TM 128
#define TN 64
#define TK 64     // two WMMA K=32 slices per stage
#define LDSS 72   // padded LDS row stride (halves): 64 + 8 pad

// issue one 16B async global->LDS copy (per-lane), tracked by ASYNCcnt
__device__ __forceinline__ void async_copy16(const void* gp, void* lds_p) {
  unsigned loff = (unsigned)(size_t)lds_p;             // LDS byte offset (low 32b)
  unsigned long long ga = (unsigned long long)(size_t)gp;
  asm volatile("global_load_async_to_lds_b128 %0, %1, off"
               :: "v"(loff), "v"(ga) : "memory");
}
__device__ __forceinline__ void wait_async0() {
  asm volatile("s_wait_asynccnt 0x0" ::: "memory");
}

// ---------------------------------------------------------------------------
// Generic GEMM:  C[M,N] = A[M,K](f16) * W[N,K](f32, ldw) with fused epilogue:
//   C += bias[n] ; C += rowvec[(m>>8)*N + n] ; C += res[m*N + n]   (optional)
//   store to outF (fp32) and/or outH (f16)
// Block: 256 threads = 8 waves. Tile 128x64x64; wave tile 32x32 -> 8 WMMA/stage.
// ---------------------------------------------------------------------------
__global__ __launch_bounds__(256) void gemm_f16_wmma(
    const _Float16* __restrict__ A, const float* __restrict__ W, int ldw,
    const float* __restrict__ bias, const float* __restrict__ res,
    const float* __restrict__ rowvec,
    float* __restrict__ outF, _Float16* __restrict__ outH,
    int M, int N, int K)
{
  __shared__ __align__(16) _Float16 As[2 * TM * LDSS];
  __shared__ __align__(16) _Float16 Bs[2 * TN * LDSS];

  const int tid  = threadIdx.x;
  const int lane = tid & 31;
  const int wid  = tid >> 5;
  const int wm   = wid & 3;   // 4 M-slabs of 32 rows
  const int wn   = wid >> 2;  // 2 N-slabs of 32 cols
  const int bm   = blockIdx.y * TM;
  const int bn   = blockIdx.x * TN;

  const int r16 = lane & 15;
  const int kb  = (lane >> 4) * 8;   // 0 or 8 (ISA 16-bit A/B lane layout)

  v8f acc[2][2] = {};

  // stage tile kt into LDS buffer b: A via async DMA, W via fp32->f16 VALU
  auto stage = [&](int kt, int b) {
    _Float16* Asb = As + b * (TM * LDSS);
    _Float16* Bsb = Bs + b * (TN * LDSS);
    const int k0 = kt * TK;
    #pragma unroll
    for (int i = 0; i < 4; ++i) {             // 1024 16B chunks of A (128x64 h)
      int c   = tid + i * 256;
      int row = c >> 3;
      int kc  = (c & 7) * 8;
      async_copy16(A + (size_t)(bm + row) * K + k0 + kc,
                   Asb + row * LDSS + kc);
    }
    #pragma unroll
    for (int i = 0; i < 2; ++i) {             // 512 8-float chunks of W (64x64 f)
      int c    = tid + i * 256;
      int nrow = c >> 3;
      int kc   = (c & 7) * 8;
      const float* wp = W + (size_t)(bn + nrow) * ldw + k0 + kc;
      _Float16* dp = Bsb + nrow * LDSS + kc;
      #pragma unroll
      for (int j = 0; j < 8; ++j) dp[j] = (_Float16)wp[j];
    }
  };

  const int nk = K / TK;
  stage(0, 0);
  wait_async0();
  __syncthreads();

  for (int kt = 0; kt < nk; ++kt) {
    const int cur = kt & 1;
    if (kt + 1 < nk) stage(kt + 1, cur ^ 1);   // overlap next stage with compute

    const _Float16* Asb = As + cur * (TM * LDSS);
    const _Float16* Bsb = Bs + cur * (TN * LDSS);

    #pragma unroll
    for (int ks = 0; ks < 2; ++ks) {           // two K=32 WMMA slices
      v16h af[2], bfg[2];
      #pragma unroll
      for (int mt = 0; mt < 2; ++mt) {
        const _Float16* base = Asb + (wm * 32 + mt * 16 + r16) * LDSS + ks * 32;
        v8h lo = *(const v8h*)(base + kb);
        v8h hi = *(const v8h*)(base + 16 + kb);
        #pragma unroll
        for (int j = 0; j < 8; ++j) { af[mt][j] = lo[j]; af[mt][j + 8] = hi[j]; }
      }
      #pragma unroll
      for (int nt = 0; nt < 2; ++nt) {
        const _Float16* base = Bsb + (wn * 32 + nt * 16 + r16) * LDSS + ks * 32;
        v8h lo = *(const v8h*)(base + kb);
        v8h hi = *(const v8h*)(base + 16 + kb);
        #pragma unroll
        for (int j = 0; j < 8; ++j) { bfg[nt][j] = lo[j]; bfg[nt][j + 8] = hi[j]; }
      }
      #pragma unroll
      for (int mt = 0; mt < 2; ++mt)
        #pragma unroll
        for (int nt = 0; nt < 2; ++nt)
          acc[mt][nt] = __builtin_amdgcn_wmma_f32_16x16x32_f16(
              false, af[mt], false, bfg[nt], (short)0, acc[mt][nt], false, false);
    }

    wait_async0();       // next-stage DMA done before publishing the buffer
    __syncthreads();
  }

  // epilogue: C/D layout => lane L: n = L&15, vgpr i: m = 8*(L>>4)+i
  const int mrow0 = bm + wm * 32;
  const int ncol0 = bn + wn * 32 + (lane & 15);
  const int mhalf = (lane >> 4) * 8;
  #pragma unroll
  for (int mt = 0; mt < 2; ++mt) {
    #pragma unroll
    for (int nt = 0; nt < 2; ++nt) {
      int n = ncol0 + nt * 16;
      #pragma unroll
      for (int i = 0; i < 8; ++i) {
        int m = mrow0 + mt * 16 + mhalf + i;
        float v = acc[mt][nt][i];
        if (bias)   v += bias[n];
        if (rowvec) v += rowvec[(size_t)(m >> 8) * N + n];
        if (res)    v += res[(size_t)m * N + n];
        size_t off = (size_t)m * N + n;
        if (outF) outF[off] = v;
        if (outH) outH[off] = (_Float16)v;
      }
    }
  }
}

// ---------------------------------------------------------------------------
__global__ void f32_to_f16_kernel(const float* __restrict__ in,
                                  _Float16* __restrict__ out, size_t n) {
  size_t i = (size_t)blockIdx.x * blockDim.x + threadIdx.x;
  if (i < n) out[i] = (_Float16)in[i];
}

// ---------------------------------------------------------------------------
__global__ __launch_bounds__(256) void rms_norm_f16(
    const float* __restrict__ x, const float* __restrict__ w,
    _Float16* __restrict__ out, int C) {
  int row = blockIdx.x;
  const float* xr = x + (size_t)row * C;
  float ss = 0.f;
  for (int c = threadIdx.x; c < C; c += 256) { float v = xr[c]; ss += v * v; }
  __shared__ float red[256];
  red[threadIdx.x] = ss; __syncthreads();
  for (int s = 128; s > 0; s >>= 1) {
    if (threadIdx.x < s) red[threadIdx.x] += red[threadIdx.x + s];
    __syncthreads();
  }
  float r = rsqrtf(red[0] / C + 1e-6f);
  for (int c = threadIdx.x; c < C; c += 256)
    out[(size_t)row * C + c] = (_Float16)(xr[c] * r * w[c]);
}

// ---------------------------------------------------------------------------
__global__ __launch_bounds__(256) void layernorm_f16(
    const float* __restrict__ x, const float* __restrict__ w,
    const float* __restrict__ b, _Float16* __restrict__ out, int C) {
  int row = blockIdx.x;
  const float* xr = x + (size_t)row * C;
  float s = 0.f, ss = 0.f;
  for (int c = threadIdx.x; c < C; c += 256) { float v = xr[c]; s += v; ss += v * v; }
  __shared__ float r1[256], r2[256];
  r1[threadIdx.x] = s; r2[threadIdx.x] = ss; __syncthreads();
  for (int st = 128; st > 0; st >>= 1) {
    if (threadIdx.x < st) { r1[threadIdx.x] += r1[threadIdx.x + st];
                            r2[threadIdx.x] += r2[threadIdx.x + st]; }
    __syncthreads();
  }
  float mu  = r1[0] / C;
  float var = r2[0] / C - mu * mu;
  float rr  = rsqrtf(var + 1e-5f);
  for (int c = threadIdx.x; c < C; c += 256)
    out[(size_t)row * C + c] = (_Float16)((xr[c] - mu) * rr * w[c] + b[c]);
}

// ---------------------------------------------------------------------------
__global__ void actvec_kernel(const float* __restrict__ actions,
                              const float* __restrict__ ap_w,
                              const float* __restrict__ ap_b,
                              const float* __restrict__ temb,
                              float* __restrict__ av) {
  int bt = blockIdx.x;            // 0..31 (= b*8 + t)
  int t  = bt & 7;                // tpos = min(t, 8) == t since t < 8
  float a0 = actions[bt * 2 + 0], a1 = actions[bt * 2 + 1];
  for (int c = threadIdx.x; c < 768; c += 256) {
    float v = ap_b[c] + temb[(size_t)t * 768 + c]
            + a0 * ap_w[(size_t)c * 770 + 768]
            + a1 * ap_w[(size_t)c * 770 + 769];
    av[(size_t)bt * 768 + c] = v;
  }
}

// ---------------------------------------------------------------------------
// Causal depthwise conv1d (K=4) + SiLU, optional per-48-group L2 norm + scale.
// In-place; grid = (channel_groups_of_48, 32 sequences), 256 threads = tokens.
// ---------------------------------------------------------------------------
__global__ __launch_bounds__(256) void conv_silu_kernel(
    float* __restrict__ data, const float* __restrict__ cw,
    int ld, int do_l2, float scale) {
  __shared__ float tile[256][49];
  int seq = blockIdx.y;
  int c0  = blockIdx.x * 48;
  size_t base = (size_t)seq * 256 * ld + c0;
  for (int idx = threadIdx.x; idx < 256 * 48; idx += 256) {
    int t = idx / 48, c = idx % 48;
    tile[t][c] = data[base + (size_t)t * ld + c];
  }
  __syncthreads();
  int t = threadIdx.x;
  float y[48];
  float ss = 0.f;
  #pragma unroll 4
  for (int c = 0; c < 48; ++c) {
    const float* w = cw + (size_t)(c0 + c) * 4;
    float acc = tile[t][c] * w[3];
    if (t >= 1) acc += tile[t - 1][c] * w[2];
    if (t >= 2) acc += tile[t - 2][c] * w[1];
    if (t >= 3) acc += tile[t - 3][c] * w[0];
    float s = acc / (1.f + expf(-acc));   // SiLU
    y[c] = s; ss += s * s;
  }
  float r = do_l2 ? rsqrtf(ss + 1e-6f) * scale : scale;
  #pragma unroll 4
  for (int c = 0; c < 48; ++c)
    data[base + (size_t)t * ld + c] = y[c] * r;
}

// ---------------------------------------------------------------------------
// beta = sigmoid(h @ wb.T); g = -exp(A_log) * softplus(h @ wa.T + dt_bias)
// ---------------------------------------------------------------------------
__global__ __launch_bounds__(384) void betag_kernel(
    const _Float16* __restrict__ h, const float* __restrict__ wb,
    const float* __restrict__ wa, const float* __restrict__ A_log,
    const float* __restrict__ dt_bias,
    float* __restrict__ beta, float* __restrict__ g) {
  int row  = blockIdx.x;
  int hd   = threadIdx.x >> 5;
  int lane = threadIdx.x & 31;
  const _Float16* hr = h + (size_t)row * 768;
  const float* wbr = wb + (size_t)hd * 768;
  const float* war = wa + (size_t)hd * 768;
  float sb = 0.f, sa = 0.f;
  for (int c = lane; c < 768; c += 32) {
    float x = (float)hr[c];
    sb += x * wbr[c]; sa += x * war[c];
  }
  for (int m = 16; m; m >>= 1) { sb += __shfl_xor(sb, m); sa += __shfl_xor(sa, m); }
  if (lane == 0) {
    float u  = sa + dt_bias[hd];
    float sp = (u > 20.f) ? u : log1pf(expf(u));
    beta[(size_t)row * 12 + hd] = 1.f / (1.f + expf(-sb));
    g[(size_t)row * 12 + hd]    = -expf(A_log[hd]) * sp;
  }
}

// ---------------------------------------------------------------------------
// Gated delta-rule scan. 1 block per (seq, head); 96 threads own v-columns of
// S[48,96] in registers; 256 sequential steps.
// ---------------------------------------------------------------------------
__global__ __launch_bounds__(96) void scan_kernel(
    const float* __restrict__ q, const float* __restrict__ k,
    const float* __restrict__ v, const float* __restrict__ beta,
    const float* __restrict__ g, float* __restrict__ o) {
  int sh  = blockIdx.x;            // seq*12 + h
  int seq = sh / 12, h = sh % 12;
  int vi  = threadIdx.x;           // 0..95
  __shared__ float kx[48], qx[48];
  float S[48];
  #pragma unroll
  for (int i = 0; i < 48; ++i) S[i] = 0.f;

  for (int t = 0; t < 256; ++t) {
    size_t row = (size_t)seq * 256 + t;
    size_t qk  = (row * 12 + h) * 48;
    if (vi < 48)      kx[vi]      = k[qk + vi];
    else              qx[vi - 48] = q[qk + (vi - 48)];
    __syncthreads();
    float gt = g[row * 12 + h];
    float bt = beta[row * 12 + h];
    float eg = expf(gt);
    float vt = v[(row * 12 + h) * 96 + vi];
    float vold = 0.f;
    #pragma unroll 8
    for (int i = 0; i < 48; ++i) { S[i] *= eg; vold += kx[i] * S[i]; }
    float dv = bt * (vt - vold);
    float on = 0.f;
    #pragma unroll 8
    for (int i = 0; i < 48; ++i) { S[i] += kx[i] * dv; on += qx[i] * S[i]; }
    o[(row * 12 + h) * 96 + vi] = on;
    __syncthreads();
  }
}

// ---------------------------------------------------------------------------
// o -> rms(o, onorm) * gate * sigmoid(gate) -> f16. 12 waves = 12 heads.
// ---------------------------------------------------------------------------
__global__ __launch_bounds__(384) void gateout_kernel(
    const float* __restrict__ o, const float* __restrict__ gate,
    const float* __restrict__ onorm, _Float16* __restrict__ out) {
  int row  = blockIdx.x;
  int h    = threadIdx.x >> 5;
  int lane = threadIdx.x & 31;
  size_t base = (size_t)row * 1152 + (size_t)h * 96;
  float v0 = o[base + lane], v1 = o[base + lane + 32], v2 = o[base + lane + 64];
  float ss = v0 * v0 + v1 * v1 + v2 * v2;
  for (int m = 16; m; m >>= 1) ss += __shfl_xor(ss, m);
  float r = rsqrtf(ss / 96.f + 1e-6f);
  float vv[3] = {v0, v1, v2};
  #pragma unroll
  for (int j = 0; j < 3; ++j) {
    int idx  = lane + j * 32;
    float gv = gate[base + idx];
    float rs = vv[j] * r * onorm[idx] * gv / (1.f + expf(-gv));
    out[base + idx] = (_Float16)rs;
  }
}

// ---------------------------------------------------------------------------
__global__ void silu_mul_kernel(const float* __restrict__ u1,
                                const float* __restrict__ u3,
                                _Float16* __restrict__ out, size_t n) {
  size_t i = (size_t)blockIdx.x * blockDim.x + threadIdx.x;
  if (i < n) {
    float a = u1[i];
    out[i] = (_Float16)((a / (1.f + expf(-a))) * u3[i]);
  }
}

// ---------------------------------------------------------------------------
static inline void launch_gemm(hipStream_t s, const _Float16* A, const float* W,
                               int ldw, const float* bias, const float* res,
                               const float* rowvec, float* outF, _Float16* outH,
                               int M, int N, int K) {
  dim3 grid(N / TN, M / TM);
  gemm_f16_wmma<<<grid, 256, 0, s>>>(A, W, ldw, bias, res, rowvec, outF, outH,
                                     M, N, K);
}

extern "C" void kernel_launch(void* const* d_in, const int* in_sizes, int n_in,
                              void* d_out, int out_size, void* d_ws, size_t ws_size,
                              hipStream_t stream) {
  // ---- inputs (setup_inputs dict order), all fp32 ----
  const float* z       = (const float*)d_in[0];   // [8192,1024]
  const float* actions = (const float*)d_in[1];   // [4,8,2]
  const float* enc_w   = (const float*)d_in[2];   // [768,1024]
  const float* enc_b   = (const float*)d_in[3];
  const float* ap_w    = (const float*)d_in[4];   // [768,770]
  const float* ap_b    = (const float*)d_in[5];
  const float* temb    = (const float*)d_in[6];   // [9,768]
  const float* ln_w    = (const float*)d_in[7];
  const float* ln_b    = (const float*)d_in[8];
  const float* dec_w   = (const float*)d_in[9];   // [1024,768]
  const float* dec_b   = (const float*)d_in[10];
  const float* n1      = (const float*)d_in[11];  // [5,768]
  const float* n2      = (const float*)d_in[12];
  const float* wq      = (const float*)d_in[13];  // [5,576,768]
  const float* cq      = (const float*)d_in[14];  // [5,576,4]
  const float* wk      = (const float*)d_in[15];
  const float* ck      = (const float*)d_in[16];
  const float* wv      = (const float*)d_in[17];  // [5,1152,768]
  const float* cv      = (const float*)d_in[18];  // [5,1152,4]
  const float* wb      = (const float*)d_in[19];  // [5,12,768]
  const float* wa      = (const float*)d_in[20];
  const float* A_log   = (const float*)d_in[21];  // [5,12]
  const float* dt_bias = (const float*)d_in[22];
  const float* wg      = (const float*)d_in[23];  // [5,1152,768]
  const float* onorm   = (const float*)d_in[24];  // [5,96]
  const float* wo      = (const float*)d_in[25];  // [5,768,1152]
  const float* w1      = (const float*)d_in[26];  // [5,2048,768]
  const float* w2      = (const float*)d_in[27];  // [5,768,2048]
  const float* w3      = (const float*)d_in[28];  // [5,2048,768]

  const int M = 8192;  // B*T*N tokens

  // ---- workspace layout (bytes) with aliasing ----
  char* ws = (char*)d_ws;
  const size_t OFF_X    = 0;                             // fp32 residual x [8192,768]
  const size_t OFF_H16  = OFF_X    + (size_t)M*768*4;    // f16 normed acts
  const size_t OFF_BUFA = OFF_H16  + (size_t)M*768*2;    // f16: z16 / mlp m16 [8192,2048]
  const size_t OFF_FO16 = OFF_BUFA + (size_t)M*2048*2;   // f16: feat16 / obf16 [8192,1152]
  const size_t OFF_Q    = OFF_FO16 + (size_t)M*1152*2;   // fp32 [8192,576]
  const size_t OFF_K    = OFF_Q    + (size_t)M*576*4;
  const size_t OFF_V    = OFF_K    + (size_t)M*576*4;    // fp32 [8192,1152] (u3 aliases V+G)
  const size_t OFF_G    = OFF_V    + (size_t)M*1152*4;   // fp32 [8192,1152]
  const size_t OFF_OU1  = OFF_G    + (size_t)M*1152*4;   // fp32 [8192,2048]: o then u1
  const size_t OFF_BETA = OFF_OU1  + (size_t)M*2048*4;
  const size_t OFF_GG   = OFF_BETA + (size_t)M*12*4;
  const size_t OFF_AV   = OFF_GG   + (size_t)M*12*4;
  const size_t TOTAL    = OFF_AV   + (size_t)32*768*4;
  if (ws_size < TOTAL) return;

  float*    x      = (float*)(ws + OFF_X);
  _Float16* h16    = (_Float16*)(ws + OFF_H16);
  _Float16* z16    = (_Float16*)(ws + OFF_BUFA);
  _Float16* m16    = (_Float16*)(ws + OFF_BUFA);
  _Float16* feat16 = (_Float16*)(ws + OFF_FO16);
  _Float16* obf16  = (_Float16*)(ws + OFF_FO16);
  float*    qb     = (float*)(ws + OFF_Q);
  float*    kbuf   = (float*)(ws + OFF_K);
  float*    vb     = (float*)(ws + OFF_V);
  float*    gb     = (float*)(ws + OFF_G);
  float*    obuf   = (float*)(ws + OFF_OU1);
  float*    u1     = (float*)(ws + OFF_OU1);
  float*    u3     = (float*)(ws + OFF_V);      // spans V+G regions (both dead)
  float*    betab  = (float*)(ws + OFF_BETA);
  float*    ggb    = (float*)(ws + OFF_GG);
  float*    actv   = (float*)(ws + OFF_AV);

  // ---- encoder + action/time embedding ----
  f32_to_f16_kernel<<<(M*1024)/256, 256, 0, stream>>>(z, z16, (size_t)M*1024);
  launch_gemm(stream, z16, enc_w, 1024, enc_b, nullptr, nullptr,
              nullptr, feat16, M, 768, 1024);
  actvec_kernel<<<32, 256, 0, stream>>>(actions, ap_w, ap_b, temb, actv);
  launch_gemm(stream, feat16, ap_w, 770, nullptr, nullptr, actv,
              x, nullptr, M, 768, 768);

  // ---- 5 GDN blocks ----
  for (int l = 0; l < 5; ++l) {
    rms_norm_f16<<<M, 256, 0, stream>>>(x, n1 + (size_t)l*768, h16, 768);

    launch_gemm(stream, h16, wq + (size_t)l*576*768, 768, nullptr, nullptr,
                nullptr, qb, nullptr, M, 576, 768);
    launch_gemm(stream, h16, wk + (size_t)l*576*768, 768, nullptr, nullptr,
                nullptr, kbuf, nullptr, M, 576, 768);
    launch_gemm(stream, h16, wv + (size_t)l*1152*768, 768, nullptr, nullptr,
                nullptr, vb, nullptr, M, 1152, 768);
    launch_gemm(stream, h16, wg + (size_t)l*1152*768, 768, nullptr, nullptr,
                nullptr, gb, nullptr, M, 1152, 768);

    conv_silu_kernel<<<dim3(12, 32), 256, 0, stream>>>(
        qb, cq + (size_t)l*576*4, 576, 1, 0.14433756729740643f); // DK^-0.5
    conv_silu_kernel<<<dim3(12, 32), 256, 0, stream>>>(
        kbuf, ck + (size_t)l*576*4, 576, 1, 1.0f);
    conv_silu_kernel<<<dim3(24, 32), 256, 0, stream>>>(
        vb, cv + (size_t)l*1152*4, 1152, 0, 1.0f);

    betag_kernel<<<M, 384, 0, stream>>>(h16, wb + (size_t)l*12*768,
        wa + (size_t)l*12*768, A_log + (size_t)l*12, dt_bias + (size_t)l*12,
        betab, ggb);

    scan_kernel<<<384, 96, 0, stream>>>(qb, kbuf, vb, betab, ggb, obuf);

    gateout_kernel<<<M, 384, 0, stream>>>(obuf, gb, onorm + (size_t)l*96, obf16);

    launch_gemm(stream, obf16, wo + (size_t)l*768*1152, 1152, nullptr, x,
                nullptr, x, nullptr, M, 768, 1152);

    // MLP
    rms_norm_f16<<<M, 256, 0, stream>>>(x, n2 + (size_t)l*768, h16, 768);
    launch_gemm(stream, h16, w1 + (size_t)l*2048*768, 768, nullptr, nullptr,
                nullptr, u1, nullptr, M, 2048, 768);
    launch_gemm(stream, h16, w3 + (size_t)l*2048*768, 768, nullptr, nullptr,
                nullptr, u3, nullptr, M, 2048, 768);
    silu_mul_kernel<<<(M*2048)/256, 256, 0, stream>>>(u1, u3, m16, (size_t)M*2048);
    launch_gemm(stream, m16, w2 + (size_t)l*768*2048, 2048, nullptr, x,
                nullptr, x, nullptr, M, 768, 2048);
  }

  // ---- final LayerNorm + decoder + residual on z ----
  layernorm_f16<<<M, 256, 0, stream>>>(x, ln_w, ln_b, h16, 768);
  launch_gemm(stream, h16, dec_w, 768, dec_b, z, nullptr,
              (float*)d_out, nullptr, M, 1024, 768);
}
